// MANNet_27676769255620
// MI455X (gfx1250) — matrix operand order repaired and checked by reading, not compiled, not run on hardware
//
#include <hip/hip_runtime.h>
#include <hip/hip_bf16.h>

// MANNet forward for MI455X (gfx1250, wave32, WMMA).
// - All GEMMs: v_wmma_f32_16x16x32_f16 (f16 in, f32 accum).
// - Weight panels staged to LDS via async global->LDS (ASYNCcnt).
// - Pairwise attentions fully fused (the (B,T,T,2H) tensor is never materialized).
// - GRU scans: persistent workgroup per direction, Whh fragments register-resident,
//   next-step gate inputs prefetched (global_prefetch_b8).

#define Bn  4
#define Tn  256
#define En  300
#define EP  320      // E padded to multiple of 32
#define Hn  128
#define H2  256
#define H3  384
#define H12 1536
#define BT  1024     // Bn*Tn

typedef _Float16 half_t;
typedef __attribute__((ext_vector_type(16))) _Float16 v16h;
typedef __attribute__((ext_vector_type(8)))  float    v8f;
typedef __attribute__((ext_vector_type(4)))  int      v4i;

// ------------------------------------------------ async global -> LDS staging

#if __has_builtin(__builtin_amdgcn_global_load_async_to_lds_b128)
#define ASYNC_TIER 1
#elif defined(__gfx1250__)
#define ASYNC_TIER 2
#else
#define ASYNC_TIER 3
#endif

__device__ __forceinline__ void g2l_b128(const half_t* g, half_t* l) {
#if ASYNC_TIER == 1
  __builtin_amdgcn_global_load_async_to_lds_b128(
      (__attribute__((address_space(1))) v4i*)g,
      (__attribute__((address_space(3))) v4i*)l, 0, 0);
#elif ASYNC_TIER == 2
  unsigned long long ga = (unsigned long long)g;
  unsigned int la = (unsigned int)(unsigned long long)l;  // LDS offset = low 32b
  asm volatile("global_load_async_to_lds_b128 %0, %1, off"
               :: "v"(la), "v"(ga) : "memory");
#else
  *(int4*)l = *(const int4*)g;
#endif
}

__device__ __forceinline__ void g2l_wait() {
#if ASYNC_TIER == 1 || ASYNC_TIER == 2
#if __has_builtin(__builtin_amdgcn_s_wait_asynccnt)
  __builtin_amdgcn_s_wait_asynccnt(0);
#else
  asm volatile("s_wait_asynccnt 0x0" ::: "memory");
#endif
#endif
}

// ---------------------------------------------------------------- fragments

// A fragment: 16 rows (m0..m0+15) x 32 K-values from row-major f16, stride lda.
// ISA 7.12.2 16-bit A 16x32: lane<16: halves0-7 = K0..7, halves8-15 = K16..23;
// lane>=16: K8..15 and K24..31 (same row = lane&15).
__device__ __forceinline__ v16h load_frag_a(const half_t* A, int lda, int m0,
                                            int k0, int lane) {
  int r = lane & 15;
  const half_t* p = A + (size_t)(m0 + r) * lda + k0 + ((lane >> 4) << 3);
  v16h a;
#pragma unroll
  for (int i = 0; i < 8; ++i) { a[i] = p[i]; a[8 + i] = p[16 + i]; }
  return a;
}

// B fragment: logical B = W^T, tile (k0..k0+31) x (n0..n0+15); W is (N,K) row-major.
// lanes 0-15 hold K k0..k0+7 and k0+16..k0+23 of column n=lane&15, lanes 16-31 hold K k0+8..k0+15 and k0+24..k0+31.
__device__ __forceinline__ v16h load_frag_b(const half_t* W, int ldw, int n0,
                                            int k0, int lane) {
  const half_t* p = W + (size_t)(n0 + (lane & 15)) * ldw + k0 + ((lane >> 4) << 3);
  v16h b;
#pragma unroll
  for (int i = 0; i < 8; ++i) { b[i] = p[i]; b[8 + i] = p[16 + i]; }
  return b;
}

__device__ __forceinline__ v8f wmma16(v16h a, v16h b, v8f c) {
  return __builtin_amdgcn_wmma_f32_16x16x32_f16(false, a, false, b, (short)0, c,
                                                false, false);
}

__device__ __forceinline__ float sigm(float x) { return 1.f / (1.f + expf(-x)); }

// ------------------------------------------------------------ prep kernels

__global__ void k_f16cvt(const float* __restrict__ src, half_t* __restrict__ dst,
                         int rows, int Ks, int Kd) {
  size_t idx = (size_t)blockIdx.x * blockDim.x + threadIdx.x;
  size_t total = (size_t)rows * Kd;
  if (idx >= total) return;
  int c = (int)(idx % Kd);
  size_t r = idx / Kd;
  dst[idx] = (c < Ks) ? (half_t)src[r * (size_t)Ks + c] : (half_t)0.f;
}

__global__ void __launch_bounds__(EP)
k_gather(const int* __restrict__ inp, const float* __restrict__ embed,
         half_t* __restrict__ X16) {
  int bt = blockIdx.x, c = threadIdx.x;
  int idx = inp[bt];
  X16[(size_t)bt * EP + c] = (c < En) ? (half_t)embed[(size_t)idx * En + c]
                                      : (half_t)0.f;
}

__global__ void k_transpose(const half_t* __restrict__ in,
                            half_t* __restrict__ out) {
  // (Bn,Tn,H2) -> (Bn,H2,Tn)
  size_t idx = (size_t)blockIdx.x * blockDim.x + threadIdx.x;
  if (idx >= (size_t)Bn * Tn * H2) return;
  int c = (int)(idx % H2);
  size_t bt = idx / H2;
  int t = (int)(bt % Tn);
  int b = (int)(bt / Tn);
  out[((size_t)b * H2 + c) * Tn + t] = in[idx];
}

__global__ void k_copy_hr(const float* __restrict__ hr, half_t* __restrict__ agg16) {
  size_t idx = (size_t)blockIdx.x * blockDim.x + threadIdx.x;
  if (idx >= (size_t)BT * H2) return;
  size_t r = idx / H2, c = idx % H2;
  agg16[r * H12 + c] = (half_t)hr[idx];
}

// ------------------------------------------------------------ generic GEMM
// C[z][m][n] = sum_k A[z][m][k] * W[z][n][k]  (+ bias[z][n])
// block = 128 (4 waves); all 4 waves share one N-tile, so the 16xK weight
// panel is staged once per block into LDS (async->LDS) and B-fragments are
// served by ds_load. grid (N/16, M/64, batch). dyn-LDS = 16*K*2 bytes.
__global__ void __launch_bounds__(128)
k_gemm_nt(const half_t* __restrict__ A, long long sA, int lda,
          const half_t* __restrict__ W, long long sW, int ldw,
          const float* __restrict__ bias, long long sBias,
          float* C, long long sC, int ldc,
          half_t* C16, long long sC16, int ldc16,
          int K) {
  extern __shared__ __align__(16) half_t wpanel[];  // 16 rows x K halves
  int z = blockIdx.z;
  int lane = threadIdx.x & 31;
  int wave = threadIdx.x >> 5;
  int m0 = (blockIdx.y * 4 + wave) * 16;
  int n0 = blockIdx.x * 16;
  const half_t* Ab = A + (long long)z * sA;
  const half_t* Wp = W + (long long)z * sW + (size_t)n0 * ldw;

  // stage the weight panel: 16*K/8 chunks of 16B
  int kc8 = K >> 3;
  for (int c = threadIdx.x; c < 16 * kc8; c += 128) {
    int row = c / kc8;
    int col = (c - row * kc8) << 3;
    g2l_b128(Wp + (size_t)row * ldw + col, &wpanel[row * K + col]);
  }
  g2l_wait();
  __syncthreads();

  v8f acc = {0.f, 0.f, 0.f, 0.f, 0.f, 0.f, 0.f, 0.f};
  for (int k0 = 0; k0 < K; k0 += 32) {
    v16h a = load_frag_a(Ab, lda, m0, k0, lane);
    v16h b = load_frag_b(wpanel, K, 0, k0, lane);
    acc = wmma16(a, b, acc);
  }
  int col = lane & 15;
  int rbase = (lane >> 4) * 8;
  float bv = bias ? bias[z * sBias + n0 + col] : 0.f;
#pragma unroll
  for (int v = 0; v < 8; ++v) {
    float val = acc[v] + bv;
    long long row = m0 + rbase + v;
    if (C)   C[z * sC + row * (long long)ldc + n0 + col] = val;
    if (C16) C16[z * sC16 + row * (long long)ldc16 + n0 + col] = (half_t)val;
  }
}

// ------------------------------------------------------------ GRU scan
// One block per direction. 24 waves: wave w owns gh columns [w*16, w*16+16).
// h kept in LDS (f16 for WMMA A, f32 for recurrence). Whh fragments preloaded
// into registers once; next-step Pre row prefetched each iteration.
__global__ void __launch_bounds__(768)
k_gru_scan(const float* __restrict__ PreL_, const float* __restrict__ PreR_,
           const half_t* __restrict__ Whh16base,
           const float* __restrict__ bhhL, const float* __restrict__ bhhR,
           float* outL, float* outR, half_t* outL16, half_t* outR16) {
  __shared__ half_t hh[16][Hn];   // batch-padded h (rows 4..15 stay zero)
  __shared__ float  gh[4][H3];
  __shared__ float  hcur[4][Hn];
  int tid = threadIdx.x, lane = tid & 31, wave = tid >> 5;
  int g = blockIdx.x >> 1, d = blockIdx.x & 1;
  const float* Pre = (g ? PreR_ : PreL_) + (size_t)d * BT * H3;
  const float* bhh = (g ? bhhR : bhhL) + d * H3;
  const half_t* Wm = Whh16base + (size_t)blockIdx.x * H3 * Hn;
  float*  out   = g ? outR : outL;
  half_t* out16 = g ? outR16 : outL16;
  int coff = d * Hn;

  for (int i = tid; i < 16 * Hn; i += 768) (&hh[0][0])[i] = (half_t)0.f;
  for (int i = tid; i < 4 * Hn; i += 768) (&hcur[0][0])[i] = 0.f;

  int n0 = wave * 16;
  v16h bf[4];
#pragma unroll
  for (int kc = 0; kc < 4; ++kc) bf[kc] = load_frag_b(Wm, Hn, n0, kc * 32, lane);
  __syncthreads();

  for (int s = 0; s < Tn; ++s) {
    int t = d ? (Tn - 1 - s) : s;
    v8f c = {0.f, 0.f, 0.f, 0.f, 0.f, 0.f, 0.f, 0.f};
#pragma unroll
    for (int kc = 0; kc < 4; ++kc) {
      v16h a = load_frag_a(&hh[0][0], Hn, 0, kc * 32, lane);
      c = wmma16(a, bf[kc], c);
    }
    if (lane < 16) {  // rows 0..3 hold batches 0..3
#pragma unroll
      for (int v = 0; v < 4; ++v)
        gh[v][n0 + lane] = c[v] + bhh[n0 + lane];
    }
    __syncthreads();
    if (tid < 512) {
      int b = tid >> 7, j = tid & 127;
      const float* pp = Pre + ((size_t)b * Tn + t) * H3;
      if (s + 1 < Tn) {  // hide next step's gate-input latency
        int t2 = d ? (Tn - 2 - s) : (s + 1);
        __builtin_prefetch(Pre + ((size_t)b * Tn + t2) * H3 + j, 0, 0);
      }
      float r = sigm(pp[j] + gh[b][j]);
      float z = sigm(pp[Hn + j] + gh[b][Hn + j]);
      float n = tanhf(pp[2 * Hn + j] + r * gh[b][2 * Hn + j]);
      float ho = hcur[b][j];
      float hn = (1.f - z) * n + z * ho;
      hcur[b][j] = hn;
      hh[b][j] = (half_t)hn;
      size_t o = ((size_t)b * Tn + t) * H2 + coff + j;
      out[o] = hn;
      out16[o] = (half_t)hn;
    }
    __syncthreads();
  }
}

// ----------------------------------------------- fused pairwise attention
// score[b][i][j] = sum_n v[n] * tanh( sum_k W[n][k] * comb(Xl[b][j][k], Hr[b][i][k]) )
// comb: op==0 -> mul, op==1 -> sub. Block 256 = 8 waves; W (128x256 f16 = 64KB)
// staged to LDS once per block; wave handles one i, a 16-wide j tile;
// 8 N-tiles x 8 K-chunks of WMMA, then v.tanh cross-lane reduction.
__global__ void __launch_bounds__(256)
k_pair_score(const half_t* __restrict__ Xl16, const half_t* __restrict__ Hr16,
             const half_t* __restrict__ W16, const float* __restrict__ v,
             float* __restrict__ Sc, int op) {
  __shared__ __align__(16) half_t wsh[Hn * H2];  // 64KB
  __shared__ float vsh[Hn];
  int tid = threadIdx.x;
  if (tid < Hn) vsh[tid] = v[tid];
  for (int c = tid; c < (Hn * H2) / 8; c += 256)
    g2l_b128(W16 + (size_t)c * 8, &wsh[c * 8]);
  g2l_wait();
  __syncthreads();

  int lane = tid & 31, wave = tid >> 5;
  int b = blockIdx.z;
  int i = blockIdx.y * 8 + wave;
  int j0 = blockIdx.x * 16;
  const half_t* xl = Xl16 + ((size_t)b * Tn + j0 + (lane & 15)) * H2;
  const half_t* hr = Hr16 + ((size_t)b * Tn + i) * H2;
  v8f acc[8];
#pragma unroll
  for (int nt = 0; nt < 8; ++nt)
    acc[nt] = (v8f){0.f, 0.f, 0.f, 0.f, 0.f, 0.f, 0.f, 0.f};
  for (int kc = 0; kc < 8; ++kc) {
    int k0 = kc * 32 + ((lane >> 4) << 3);
    v16h a;
#pragma unroll
    for (int q = 0; q < 8; ++q) {
      half_t l0 = xl[k0 + q],      r0 = hr[k0 + q];
      half_t l1 = xl[k0 + 16 + q], r1 = hr[k0 + 16 + q];
      a[q]     = op ? (half_t)(l0 - r0) : (half_t)(l0 * r0);
      a[8 + q] = op ? (half_t)(l1 - r1) : (half_t)(l1 * r1);
    }
#pragma unroll
    for (int nt = 0; nt < 8; ++nt) {
      v16h bfr = load_frag_b(wsh, H2, nt * 16, kc * 32, lane);
      acc[nt] = wmma16(a, bfr, acc[nt]);
    }
  }
  int col = lane & 15, rbase = (lane >> 4) * 8;
  float part[8];
#pragma unroll
  for (int vv = 0; vv < 8; ++vv) {
    float s = 0.f;
#pragma unroll
    for (int nt = 0; nt < 8; ++nt) s += vsh[nt * 16 + col] * tanhf(acc[nt][vv]);
#pragma unroll
    for (int m = 1; m <= 8; m <<= 1) s += __shfl_xor(s, m, 32);
    part[vv] = s;
  }
  if ((lane & 15) == 0) {
    float* dst = Sc + ((size_t)b * Tn + i) * Tn + j0 + rbase;
#pragma unroll
    for (int vv = 0; vv < 8; ++vv) dst[vv] = part[vv];
  }
}

// additive co-attention score: vc . tanh(HLc1[b,j,:] + HRc2[b,i,:])
__global__ void __launch_bounds__(256)
k_add_score(const float* __restrict__ HLc1, const float* __restrict__ HRc2,
            const float* __restrict__ vc, float* __restrict__ Sc) {
  __shared__ float ri[Hn], vsh[Hn];
  int b = blockIdx.y, i = blockIdx.x, j = threadIdx.x;
  if (j < Hn) {
    ri[j] = HRc2[((size_t)b * Tn + i) * Hn + j];
    vsh[j] = vc[j];
  }
  __syncthreads();
  const float* lj = HLc1 + ((size_t)b * Tn + j) * Hn;
  float s = 0.f;
#pragma unroll 4
  for (int k = 0; k < Hn; ++k) s += vsh[k] * tanhf(lj[k] + ri[k]);
  Sc[((size_t)b * Tn + i) * Tn + j] = s;
}

// ------------------------------------------------------------ small kernels

__global__ void __launch_bounds__(256)
k_softmax_rows(float* __restrict__ X, half_t* __restrict__ X16, int rowlen) {
  int row = blockIdx.x, tid = threadIdx.x;
  float* xr = X + (size_t)row * rowlen;
  float v = xr[tid];
  __shared__ float red[8];
  float m = v;
#pragma unroll
  for (int s = 16; s >= 1; s >>= 1) m = fmaxf(m, __shfl_xor(m, s, 32));
  if ((tid & 31) == 0) red[tid >> 5] = m;
  __syncthreads();
  if (tid == 0) {
    float t = red[0];
    for (int i = 1; i < 8; ++i) t = fmaxf(t, red[i]);
    red[0] = t;
  }
  __syncthreads();
  float mx = red[0];
  __syncthreads();
  float e = expf(v - mx);
  float s = e;
#pragma unroll
  for (int k = 16; k >= 1; k >>= 1) s += __shfl_xor(s, k, 32);
  if ((tid & 31) == 0) red[tid >> 5] = s;
  __syncthreads();
  if (tid == 0) {
    float t = 0.f;
    for (int i = 0; i < 8; ++i) t += red[i];
    red[0] = t;
  }
  __syncthreads();
  float p = e / red[0];
  xr[tid] = p;
  if (X16) X16[(size_t)row * rowlen + tid] = (half_t)p;
}

__global__ void __launch_bounds__(128)
k_vdot_tanh(const float* __restrict__ Xin, const float* __restrict__ v,
            float* __restrict__ out) {
  int row = blockIdx.x, tid = threadIdx.x;
  float t = v[tid] * tanhf(Xin[(size_t)row * Hn + tid]);
#pragma unroll
  for (int s = 16; s >= 1; s >>= 1) t += __shfl_xor(t, s, 32);
  __shared__ float red[4];
  if ((tid & 31) == 0) red[tid >> 5] = t;
  __syncthreads();
  if (tid == 0) out[row] = red[0] + red[1] + red[2] + red[3];
}

__global__ void __launch_bounds__(256)
k_rl(const float* __restrict__ sjP, const float* __restrict__ hl,
     float* __restrict__ rl) {
  int b = blockIdx.x, c = threadIdx.x;
  float s = 0.f;
  for (int t = 0; t < Tn; ++t)
    s += sjP[b * Tn + t] * hl[((size_t)b * Tn + t) * H2 + c];
  rl[b * H2 + c] = s;
}

__global__ void __launch_bounds__(128)
k_rlc2(const float* __restrict__ Wc2, const float* __restrict__ rl,
       float* __restrict__ rlc2) {
  int b = blockIdx.x, k = threadIdx.x;
  float s = 0.f;
  for (int c = 0; c < H2; ++c) s += Wc2[k * H2 + c] * rl[b * H2 + c];
  rlc2[b * Hn + k] = s;
}

__global__ void __launch_bounds__(256)
k_u2(const float* __restrict__ ARc1, const float* __restrict__ rlc2,
     const float* __restrict__ vc, float* __restrict__ u2) {
  __shared__ float rsh[Hn], vsh[Hn];
  int b = blockIdx.x, t = threadIdx.x;
  if (t < Hn) { rsh[t] = rlc2[b * Hn + t]; vsh[t] = vc[t]; }
  __syncthreads();
  const float* ap = ARc1 + ((size_t)b * Tn + t) * Hn;
  float s = 0.f;
#pragma unroll 4
  for (int k = 0; k < Hn; ++k) s += vsh[k] * (ap[k] + rsh[k]);
  u2[b * Tn + t] = s;
}

__global__ void __launch_bounds__(256)
k_final(const float* __restrict__ sj2, const float* __restrict__ ar,
        const float* __restrict__ Wpred, float* __restrict__ out) {
  __shared__ float rr[H2];
  int b = blockIdx.x, c = threadIdx.x;
  float s = 0.f;
  for (int t = 0; t < Tn; ++t)
    s += sj2[b * Tn + t] * ar[((size_t)b * Tn + t) * H2 + c];
  rr[c] = s;
  __syncthreads();
  if (c < 2) {
    float s2 = 0.f;
    for (int k = 0; k < H2; ++k) s2 += Wpred[c * H2 + k] * rr[k];
    out[b * 2 + c] = sigm(s2);
  }
}

// ------------------------------------------------------------------- host

extern "C" void kernel_launch(void* const* d_in, const int* in_sizes, int n_in,
                              void* d_out, int out_size, void* d_ws, size_t ws_size,
                              hipStream_t stream) {
  const int*   inputs = (const int*)d_in[0];
  const float* embed  = (const float*)d_in[1];
  const float* l_Wih = (const float*)d_in[2];
  const float* l_Whh = (const float*)d_in[3];
  const float* l_bih = (const float*)d_in[4];
  const float* l_bhh = (const float*)d_in[5];
  const float* r_Wih = (const float*)d_in[6];
  const float* r_Whh = (const float*)d_in[7];
  const float* r_bih = (const float*)d_in[8];
  const float* r_bhh = (const float*)d_in[9];
  const float* a_Wih = (const float*)d_in[10];
  const float* a_Whh = (const float*)d_in[11];
  const float* a_bih = (const float*)d_in[12];
  const float* a_bhh = (const float*)d_in[13];
  const float* Wc1 = (const float*)d_in[14];
  const float* Wc2 = (const float*)d_in[15];
  const float* vc  = (const float*)d_in[16];
  const float* Wb  = (const float*)d_in[17];
  const float* Wd  = (const float*)d_in[18];
  const float* vd  = (const float*)d_in[19];
  const float* Wm  = (const float*)d_in[20];
  const float* vm  = (const float*)d_in[21];
  const float* Ws  = (const float*)d_in[22];
  const float* vs  = (const float*)d_in[23];
  const float* Wp  = (const float*)d_in[24];
  const float* vp  = (const float*)d_in[25];
  const float* Wpred = (const float*)d_in[26];

  char* ws = (char*)d_ws;
  size_t off = 0;
  auto alloc = [&](size_t bytes) -> void* {
    off = (off + 255) & ~(size_t)255;
    void* p = ws + off;
    off += bytes;
    return p;
  };

  half_t* X16     = (half_t*)alloc((size_t)BT * EP * 2);
  half_t* WihL16  = (half_t*)alloc((size_t)2 * H3 * EP * 2);
  half_t* WihR16  = (half_t*)alloc((size_t)2 * H3 * EP * 2);
  half_t* WihA16  = (half_t*)alloc((size_t)2 * H3 * H12 * 2);
  half_t* Whh16   = (half_t*)alloc((size_t)6 * H3 * Hn * 2);
  half_t* Wc116   = (half_t*)alloc((size_t)Hn * H2 * 2);
  half_t* Wc216   = (half_t*)alloc((size_t)Hn * H2 * 2);
  half_t* Wd16    = (half_t*)alloc((size_t)Hn * H2 * 2);
  half_t* Wm16    = (half_t*)alloc((size_t)Hn * H2 * 2);
  half_t* Ws16    = (half_t*)alloc((size_t)Hn * H2 * 2);
  half_t* Wp16    = (half_t*)alloc((size_t)Hn * H2 * 2);
  half_t* Wb16    = (half_t*)alloc((size_t)H2 * H2 * 2);
  float*  PreL    = (float*)alloc((size_t)2 * BT * H3 * 4);
  float*  PreR    = (float*)alloc((size_t)2 * BT * H3 * 4);
  float*  PreA    = (float*)alloc((size_t)2 * BT * H3 * 4);
  float*  hl      = (float*)alloc((size_t)BT * H2 * 4);
  float*  hr      = (float*)alloc((size_t)BT * H2 * 4);
  float*  arr     = (float*)alloc((size_t)BT * H2 * 4);
  half_t* hl16    = (half_t*)alloc((size_t)BT * H2 * 2);
  half_t* hr16    = (half_t*)alloc((size_t)BT * H2 * 2);
  half_t* ar16    = (half_t*)alloc((size_t)BT * H2 * 2);
  half_t* hlT16   = (half_t*)alloc((size_t)BT * H2 * 2);
  half_t* hrT16   = (half_t*)alloc((size_t)BT * H2 * 2);
  float*  HLc1    = (float*)alloc((size_t)BT * Hn * 4);
  float*  HRc2    = (float*)alloc((size_t)BT * Hn * 4);
  float*  HLp     = (float*)alloc((size_t)BT * Hn * 4);
  float*  ARc1    = (float*)alloc((size_t)BT * Hn * 4);
  half_t* HLb16   = (half_t*)alloc((size_t)BT * H2 * 2);
  float*  Sc      = (float*)alloc((size_t)Bn * Tn * Tn * 4);
  half_t* P16c    = (half_t*)alloc((size_t)Bn * Tn * Tn * 2);
  half_t* P16b    = (half_t*)alloc((size_t)Bn * Tn * Tn * 2);
  half_t* P16d    = (half_t*)alloc((size_t)Bn * Tn * Tn * 2);
  half_t* P16m    = (half_t*)alloc((size_t)Bn * Tn * Tn * 2);
  half_t* P16s    = (half_t*)alloc((size_t)Bn * Tn * Tn * 2);
  half_t* agg16   = (half_t*)alloc((size_t)BT * H12 * 2);
  float*  u1      = (float*)alloc((size_t)Bn * Tn * 4);
  float*  u2      = (float*)alloc((size_t)Bn * Tn * 4);
  float*  rl      = (float*)alloc((size_t)Bn * H2 * 4);
  float*  rlc2    = (float*)alloc((size_t)Bn * Hn * 4);
  (void)ws_size; (void)in_sizes; (void)n_in; (void)out_size;

  auto cvt = [&](const float* s, half_t* d, int rows, int Ks, int Kd) {
    size_t total = (size_t)rows * Kd;
    k_f16cvt<<<(unsigned)((total + 255) / 256), 256, 0, stream>>>(s, d, rows, Ks, Kd);
  };
  auto gemmLds = [](int K) { return (size_t)16 * K * 2; };

  // ---- weight conversion to f16 (K zero-padded where needed)
  cvt(l_Wih, WihL16, 2 * H3, En, EP);
  cvt(r_Wih, WihR16, 2 * H3, En, EP);
  cvt(a_Wih, WihA16, 2 * H3, H12, H12);
  cvt(l_Whh, Whh16 + 0 * (size_t)2 * H3 * Hn, 2 * H3, Hn, Hn);
  cvt(r_Whh, Whh16 + 1 * (size_t)2 * H3 * Hn, 2 * H3, Hn, Hn);
  cvt(a_Whh, Whh16 + 2 * (size_t)2 * H3 * Hn, 2 * H3, Hn, Hn);
  cvt(Wc1, Wc116, Hn, H2, H2);
  cvt(Wc2, Wc216, Hn, H2, H2);
  cvt(Wd, Wd16, Hn, H2, H2);
  cvt(Wm, Wm16, Hn, H2, H2);
  cvt(Ws, Ws16, Hn, H2, H2);
  cvt(Wp, Wp16, Hn, H2, H2);
  cvt(Wb, Wb16, H2, H2, H2);

  // ---- embedding gather (f16, padded)
  k_gather<<<BT, EP, 0, stream>>>(inputs, embed, X16);

  // ---- GRU input projections: Pre = X @ Wih^T + bih  (batched over direction)
  {
    dim3 g(H3 / 16, BT / 64, 2);
    k_gemm_nt<<<g, 128, gemmLds(EP), stream>>>(
        X16, 0, EP, WihL16, (long long)H3 * EP, EP, l_bih, H3,
        PreL, (long long)BT * H3, H3, nullptr, 0, 0, EP);
    k_gemm_nt<<<g, 128, gemmLds(EP), stream>>>(
        X16, 0, EP, WihR16, (long long)H3 * EP, EP, r_bih, H3,
        PreR, (long long)BT * H3, H3, nullptr, 0, 0, EP);
  }

  // ---- l/r bidirectional GRU scans (4 persistent workgroups)
  k_gru_scan<<<4, 768, 0, stream>>>(PreL, PreR, Whh16, l_bhh, r_bhh,
                                    hl, hr, hl16, hr16);

  // ---- transposed f16 copies for the P@hl / P@hr GEMMs
  {
    size_t total = (size_t)Bn * Tn * H2;
    unsigned nb = (unsigned)((total + 255) / 256);
    k_transpose<<<nb, 256, 0, stream>>>(hl16, hlT16);
    k_transpose<<<nb, 256, 0, stream>>>(hr16, hrT16);
  }

  // ---- attention projections
  k_gemm_nt<<<dim3(Hn / 16, BT / 64, 1), 128, gemmLds(H2), stream>>>(
      hl16, 0, H2, Wc116, 0, H2, nullptr, 0, HLc1, 0, Hn, nullptr, 0, 0, H2);
  k_gemm_nt<<<dim3(Hn / 16, BT / 64, 1), 128, gemmLds(H2), stream>>>(
      hr16, 0, H2, Wc216, 0, H2, nullptr, 0, HRc2, 0, Hn, nullptr, 0, 0, H2);
  k_gemm_nt<<<dim3(H2 / 16, BT / 64, 1), 128, gemmLds(H2), stream>>>(
      hl16, 0, H2, Wb16, 0, H2, nullptr, 0, nullptr, 0, 0, HLb16, 0, H2, H2);
  k_gemm_nt<<<dim3(Hn / 16, BT / 64, 1), 128, gemmLds(H2), stream>>>(
      hl16, 0, H2, Wp16, 0, H2, nullptr, 0, HLp, 0, Hn, nullptr, 0, 0, H2);

  // ---- scores + softmax (Sc reused across the 5 attentions)
  k_add_score<<<dim3(Tn, Bn), 256, 0, stream>>>(HLc1, HRc2, vc, Sc);
  k_softmax_rows<<<Bn * Tn, 256, 0, stream>>>(Sc, P16c, Tn);

  k_gemm_nt<<<dim3(Tn / 16, Tn / 64, Bn), 128, gemmLds(H2), stream>>>(
      hr16, (long long)Tn * H2, H2, HLb16, (long long)Tn * H2, H2, nullptr, 0,
      Sc, (long long)Tn * Tn, Tn, nullptr, 0, 0, H2);
  k_softmax_rows<<<Bn * Tn, 256, 0, stream>>>(Sc, P16b, Tn);

  k_pair_score<<<dim3(Tn / 16, Tn / 8, Bn), 256, 0, stream>>>(hl16, hr16, Wd16, vd, Sc, 0);
  k_softmax_rows<<<Bn * Tn, 256, 0, stream>>>(Sc, P16d, Tn);

  k_pair_score<<<dim3(Tn / 16, Tn / 8, Bn), 256, 0, stream>>>(hl16, hr16, Wm16, vm, Sc, 1);
  k_softmax_rows<<<Bn * Tn, 256, 0, stream>>>(Sc, P16m, Tn);

  k_pair_score<<<dim3(Tn / 16, Tn / 8, Bn), 256, 0, stream>>>(hr16, hr16, Ws16, vs, Sc, 0);
  k_softmax_rows<<<Bn * Tn, 256, 0, stream>>>(Sc, P16s, Tn);

  // ---- aggregation agg = [hr, pts, ptc, ptd, ptb, ptm] (f16, ldc = 12H)
  {
    size_t total = (size_t)BT * H2;
    k_copy_hr<<<(unsigned)((total + 255) / 256), 256, 0, stream>>>(hr, agg16);
  }
  auto ptX = [&](half_t* P16, half_t* WT, int colOff) {
    k_gemm_nt<<<dim3(H2 / 16, Tn / 64, Bn), 128, gemmLds(Tn), stream>>>(
        P16, (long long)Tn * Tn, Tn, WT, (long long)H2 * Tn, Tn, nullptr, 0,
        nullptr, 0, 0, agg16 + colOff, (long long)Tn * H12, H12, Tn);
  };
  ptX(P16s, hrT16, 1 * H2);  // pts
  ptX(P16c, hlT16, 2 * H2);  // ptc
  ptX(P16d, hlT16, 3 * H2);  // ptd
  ptX(P16b, hlT16, 4 * H2);  // ptb
  ptX(P16m, hlT16, 5 * H2);  // ptm

  // ---- aggregation GRU
  k_gemm_nt<<<dim3(H3 / 16, BT / 64, 2), 128, gemmLds(H12), stream>>>(
      agg16, 0, H12, WihA16, (long long)H3 * H12, H12, a_bih, H3,
      PreA, (long long)BT * H3, H3, nullptr, 0, 0, H12);
  k_gru_scan<<<2, 768, 0, stream>>>(PreA, PreA, Whh16 + (size_t)4 * H3 * Hn,
                                    a_bhh, a_bhh, arr, arr, ar16, ar16);

  // ---- head
  k_gemm_nt<<<dim3(Hn / 16, BT / 64, 1), 128, gemmLds(H2), stream>>>(
      ar16, 0, H2, Wc116, 0, H2, nullptr, 0, ARc1, 0, Hn, nullptr, 0, 0, H2);

  k_vdot_tanh<<<BT, 128, 0, stream>>>(HLp, vp, u1);
  k_softmax_rows<<<Bn, 256, 0, stream>>>(u1, nullptr, Tn);
  k_rl<<<Bn, 256, 0, stream>>>(u1, hl, rl);
  k_rlc2<<<Bn, 128, 0, stream>>>(Wc2, rl, rlc2);
  k_u2<<<Bn, 256, 0, stream>>>(ARc1, rlc2, vc, u2);
  k_softmax_rows<<<Bn, 256, 0, stream>>>(u2, nullptr, Tn);
  k_final<<<Bn, 256, 0, stream>>>(u2, arr, Wpred, (float*)d_out);
}
